// BP_HDFNN_70042326663684
// MI455X (gfx1250) — matrix operand order repaired and checked by reading, not compiled, not run on hardware
//
#include <hip/hip_runtime.h>

typedef __attribute__((ext_vector_type(16))) _Float16 v16h;
typedef __attribute__((ext_vector_type(8)))  float    v8f;

#define ABDIM 32   // A*B = 8*4
#define RDIM  16
#define FDIM  32
#define NDIM  4096
#define ADIM  8
#define BDIM  4

// ds_swizzle group-of-32 immediate: SWAPX16 (xor_mask=0x10, or=0, and=0x1f)
#define SWIZZLE_SWAPX16 0x401F

__device__ __forceinline__ float swap16_add(float v) {
    // v + (v from lane ^ 16): single ds_swizzle_b32, no lane-index math
    int o = __builtin_amdgcn_ds_swizzle(__float_as_int(v), SWIZZLE_SWAPX16);
    return v + __int_as_float(o);
}

// ---------------------------------------------------------------------------
// Kernel 1: transform (mu, sigma, w3) into WMMA-A-layout f16 matrices + row
// constants.  One wave (32 lanes) per (a,b) pair.
//
// A-operand packing used here (16x32, contiguous k, same as B): lane L (L<16) holds row
// M=L with K = 0..15 in element order e -> k = e; lanes
// 16..31 hold the same rows with K shifted by +16.
// ---------------------------------------------------------------------------
__global__ void prep_kernel(const float* __restrict__ mu,
                            const float* __restrict__ sg,
                            const float* __restrict__ w3,
                            _Float16* __restrict__ ahW,
                            _Float16* __restrict__ ahM1,
                            _Float16* __restrict__ ahW3,
                            float* __restrict__ cvec,
                            float* __restrict__ bias3)
{
    const int ab = blockIdx.x;          // 0..31
    const int ln = threadIdx.x;         // 0..31
    const int r  = ln & 15;
    const int kshift = (ln >> 4) << 4;  // 0 or 16

    const float* murow = mu + (size_t)(ab * RDIM + r) * FDIM;
    const float* sgrow = sg + (size_t)(ab * RDIM + r) * FDIM;
    const float* w3row = w3 + (size_t)(ab * RDIM + r) * (FDIM + 1);

    v16h wv, m1v, w3v;
    float csum = 0.0f;
#pragma unroll
    for (int e = 0; e < 16; ++e) {
        const int k = e + kshift;             // A-layout K index
        const float m   = murow[k];
        const float s   = sgrow[k];
        const float inv = 0.5f / (s * s);     // 1/(2*sigma^2)
        wv[e]  = (_Float16)inv;
        m1v[e] = (_Float16)(-2.0f * m * inv); // -mu/sigma^2
        w3v[e] = (_Float16)w3row[k];
        csum  += m * m * inv;                 // mu^2/(2*sigma^2)
    }
    // this lane covered 16 of the 32 K values of row r; partner lane (xor 16)
    // covered the other 16 -> full row constant.
    csum = swap16_add(csum);

    const size_t base = ((size_t)ab * 32 + ln) * 16;
    *(v16h*)(ahW  + base) = wv;
    *(v16h*)(ahM1 + base) = m1v;
    *(v16h*)(ahW3 + base) = w3v;
    if (ln < 16) {
        cvec [ab * RDIM + r] = csum;
        bias3[ab * RDIM + r] = w3row[FDIM];   // w3[..., -1]
    }
}

// ---------------------------------------------------------------------------
// Kernel 2: fused fuzzy-rule + consequent GEMMs via WMMA.
// One wave per (ab, 16-column tile of N).  3x v_wmma_f32_16x16x32_f16.
//
// B-operand order used here (32x16, contiguous k, same as A): lane n (n<16) holds column n, K=0..15;
// lane n+16 holds column n, K=16..31 -> contiguous 64B f32 load per lane.
// C/D layout: lanes 0..15 VGPR i <-> M=i; lanes 16..31 VGPR i <-> M=i+8.
// ---------------------------------------------------------------------------
__global__ void fused_kernel(const float* __restrict__ data,
                             const _Float16* __restrict__ ahW,
                             const _Float16* __restrict__ ahM1,
                             const _Float16* __restrict__ ahW3,
                             const float* __restrict__ cvec,
                             const float* __restrict__ bias3,
                             float* __restrict__ tsk)
{
    const int tile = blockIdx.x;        // 0..255
    const int ab   = blockIdx.y;        // 0..31
    const int ln   = threadIdx.x;       // 0..31
    const int col  = ln & 15;
    const int kbase = (ln >> 4) << 4;   // 0 or 16
    const int n    = tile * 16 + col;

    // ---- load B operand: 16 contiguous floats of data[ab, n, kbase..+15]
    const float* dp = data + ((size_t)ab * NDIM + n) * FDIM + kbase;
    const float4 d0 = ((const float4*)dp)[0];
    const float4 d1 = ((const float4*)dp)[1];
    const float4 d2 = ((const float4*)dp)[2];
    const float4 d3 = ((const float4*)dp)[3];
    const float xf[16] = {d0.x, d0.y, d0.z, d0.w, d1.x, d1.y, d1.z, d1.w,
                          d2.x, d2.y, d2.z, d2.w, d3.x, d3.y, d3.z, d3.w};
    v16h xh, x2h;
#pragma unroll
    for (int e = 0; e < 16; ++e) {
        const float v = xf[e];
        xh[e]  = (_Float16)v;
        x2h[e] = (_Float16)(v * v);
    }

    // ---- load A operands (pre-laid-out per lane, 32B each)
    const size_t abase = ((size_t)ab * 32 + ln) * 16;
    const v16h aW = *(const v16h*)(ahW  + abase);
    const v16h aM = *(const v16h*)(ahM1 + abase);
    const v16h a3 = *(const v16h*)(ahW3 + abase);

    // ---- preload accumulators with per-row constants
    const int rbase = (ln >> 4) << 3;   // 0 or 8
    v8f accS, accQ;
#pragma unroll
    for (int i = 0; i < 8; ++i) {
        accS[i] = cvec [ab * RDIM + rbase + i];
        accQ[i] = bias3[ab * RDIM + rbase + i];
    }

    // S = x^2 * (1/2s^2)  +  x * (-mu/s^2)  +  mu^2/(2s^2)
    accS = __builtin_amdgcn_wmma_f32_16x16x32_f16(false, aW, false, x2h,
                                                  (short)0, accS, false, false);
    accS = __builtin_amdgcn_wmma_f32_16x16x32_f16(false, aM, false, xh,
                                                  (short)0, accS, false, false);
    // conq = w3 * x + bias
    accQ = __builtin_amdgcn_wmma_f32_16x16x32_f16(false, a3, false, xh,
                                                  (short)0, accQ, false, false);

    // ---- rule = exp(-S); reduce over the 16 rules of this column
    float s = 0.0f, t = 0.0f;
#pragma unroll
    for (int i = 0; i < 8; ++i) {
        const float rl = __expf(-accS[i]);
        s += rl;
        t += rl * accQ[i];
    }
    // rows 0..7 live in lanes 0..15, rows 8..15 in lanes 16..31 (same column)
    s = swap16_add(s);
    t = swap16_add(t);

    if (ln < 16) tsk[(size_t)ab * NDIM + n] = t / s;
}

// ---------------------------------------------------------------------------
// Kernel 3: out[a,n] = sum_b w5[a,b]*tsk[a,b,n] + w5[a,B]; softmax over a.
// ---------------------------------------------------------------------------
__global__ void finalize_kernel(const float* __restrict__ tsk,
                                const float* __restrict__ w5,
                                float* __restrict__ out)
{
    const int n = blockIdx.x * blockDim.x + threadIdx.x;
    if (n >= NDIM) return;

    float o[ADIM];
    float mx = -3.402823466e38f;
#pragma unroll
    for (int a = 0; a < ADIM; ++a) {
        float acc = w5[a * (BDIM + 1) + BDIM];
#pragma unroll
        for (int b = 0; b < BDIM; ++b)
            acc += w5[a * (BDIM + 1) + b] * tsk[(size_t)(a * BDIM + b) * NDIM + n];
        o[a] = acc;
        mx = fmaxf(mx, acc);
    }
    float sum = 0.0f;
#pragma unroll
    for (int a = 0; a < ADIM; ++a) { o[a] = __expf(o[a] - mx); sum += o[a]; }
    const float inv = 1.0f / sum;
#pragma unroll
    for (int a = 0; a < ADIM; ++a) out[(size_t)a * NDIM + n] = o[a] * inv;
}

// ---------------------------------------------------------------------------
extern "C" void kernel_launch(void* const* d_in, const int* in_sizes, int n_in,
                              void* d_out, int out_size, void* d_ws, size_t ws_size,
                              hipStream_t stream)
{
    (void)in_sizes; (void)n_in; (void)out_size; (void)ws_size;

    const float* data = (const float*)d_in[0];   // (8,4,4096,32)
    const float* mu   = (const float*)d_in[1];   // (8,4,16,32)
    const float* sg   = (const float*)d_in[2];   // (8,4,16,32)
    const float* w3   = (const float*)d_in[3];   // (8,4,16,33)
    const float* w5   = (const float*)d_in[4];   // (8,5)
    float* out        = (float*)d_out;           // (8,4096)

    // workspace layout (bytes)
    char* ws = (char*)d_ws;
    _Float16* ahW   = (_Float16*)(ws);                         // 32 KB
    _Float16* ahM1  = (_Float16*)(ws + 32 * 1024);             // 32 KB
    _Float16* ahW3  = (_Float16*)(ws + 64 * 1024);             // 32 KB
    float*    cvec  = (float*)   (ws + 96 * 1024);             //  2 KB
    float*    bias3 = (float*)   (ws + 98 * 1024);             //  2 KB
    float*    tsk   = (float*)   (ws + 128 * 1024);            // 512 KB

    prep_kernel<<<dim3(ABDIM), dim3(32), 0, stream>>>(mu, sg, w3,
                                                      ahW, ahM1, ahW3,
                                                      cvec, bias3);

    fused_kernel<<<dim3(NDIM / 16, ABDIM), dim3(32), 0, stream>>>(
        data, ahW, ahM1, ahW3, cvec, bias3, tsk);

    finalize_kernel<<<dim3(NDIM / 256), dim3(256), 0, stream>>>(tsk, w5, out);
}
